// HashRoutedSSMLayer_55301998903669
// MI455X (gfx1250) — compile-verified, matrix operand order checked by reading, not run on hardware
//
#include <hip/hip_runtime.h>
#include <hip/hip_bf16.h>
#include <stdint.h>

#define BB 8
#define SS 2048
#define DD 1024
#define NN 256
#define HH 512
#define EE 8
#define BS (BB * SS)                 // 16384 tokens
#define TOK_TILE 32
#define TILE_MAX (BS / TOK_TILE + EE)  // worst-case 32-token tiles over all experts

typedef __bf16 bf16;
typedef __attribute__((ext_vector_type(16))) __bf16 v16bf;
typedef __attribute__((ext_vector_type(8)))  __bf16 v8bf;
typedef __attribute__((ext_vector_type(8)))  float   v8f;

__device__ __forceinline__ v8f vzero8() {
  v8f v;
#pragma unroll
  for (int i = 0; i < 8; ++i) v[i] = 0.f;
  return v;
}

__device__ __forceinline__ float sigmoidf_(float x) { return 1.f / (1.f + __expf(-x)); }

// D = A(16xK, rows=M striped per lane) x B(Kx16, cols=N striped per lane) + C
__device__ __forceinline__ v8f wmma_bf16(v16bf a, v16bf b, v8f c) {
  return __builtin_amdgcn_wmma_f32_16x16x32_bf16(false, a, false, b, (short)0, c, false, false);
}

// Load one 16x32 bf16 fragment from a row-major tile (leading dim ld, elements).
// Per ISA 7.12.2: lane (r = lane&15, h = lane>>4) holds K = k0+h*8+{0..7} and k0+16+h*8+{0..7} of row r.
__device__ __forceinline__ v16bf load_frag(const bf16* tile, int ld, int k0, int lane) {
  const int r = lane & 15;
  const int h = (lane >> 4) & 1;
  const bf16* p = tile + (size_t)r * ld + k0 + h * 8;
  v8bf c0 = *(const v8bf*)(p);
  v8bf c1 = *(const v8bf*)(p + 16);
  v16bf f;
#pragma unroll
  for (int i = 0; i < 8; ++i) { f[i] = c0[i]; f[i + 8] = c1[i]; }
  return f;
}

struct TileInfo { int e, listBase, listEnd; };

__device__ __forceinline__ bool tile_info(const int* offsets, const int* tileOff,
                                          const int* totalTiles, TileInfo& ti) {
  int tile = blockIdx.x;
  if (tile >= *totalTiles) return false;
  int e = 0;
#pragma unroll
  for (int k = 1; k < EE; ++k) if (tile >= tileOff[k]) e = k;
  ti.e = e;
  int local = tile - tileOff[e];
  ti.listBase = offsets[e] + local * TOK_TILE;
  ti.listEnd = offsets[e + 1];
  return true;
}

// Cooperative gather of TOK_TILE token rows (ROW_ELEMS bf16 each) into LDS; invalid -> 0.
template <int ROW_ELEMS>
__device__ __forceinline__ void gather_rows(const bf16* src_base, bf16* dst, const int* tl) {
  constexpr int CPR = ROW_ELEMS / 8;          // 16B chunks per row
  for (int idx = threadIdx.x; idx < TOK_TILE * CPR; idx += 256) {
    int row = idx / CPR;
    int c = idx % CPR;
    int tok = tl[row];
    int tsafe = tok < 0 ? 0 : tok;
    const uint4* s = (const uint4*)(src_base + (size_t)tsafe * ROW_ELEMS);
    uint4* d = (uint4*)(dst + (size_t)row * ROW_ELEMS);
    uint4 v;
    if (tok >= 0) v = s[c];
    else { v.x = 0u; v.y = 0u; v.z = 0u; v.w = 0u; }
    d[c] = v;
  }
}

// ---------------- small prep kernels ----------------

__global__ void k_init(int* counts) {
  if (threadIdx.x < EE) counts[threadIdx.x] = 0;
}

__global__ void k_f2bf(const float* __restrict__ in, bf16* __restrict__ out, int n) {
  int i = blockIdx.x * blockDim.x + threadIdx.x;
  if (i < n) out[i] = (bf16)in[i];
}

__global__ void k_route(const int* __restrict__ tok, int* __restrict__ routes,
                        int* __restrict__ counts) {
  int i = blockIdx.x * blockDim.x + threadIdx.x;
  if (i >= BS) return;
  unsigned x = (unsigned)tok[i];
  x ^= x >> 16; x *= 2246822507u;
  x ^= x >> 13; x *= 3266489909u;
  x ^= x >> 16;
  int r = (int)(x & 7u);   // % 8
  routes[i] = r;
  atomicAdd(&counts[r], 1);
}

__global__ void k_offsets(const int* __restrict__ counts, int* __restrict__ offsets,
                          int* __restrict__ tileOff, int* __restrict__ cursors,
                          int* __restrict__ totalTiles) {
  if (threadIdx.x != 0 || blockIdx.x != 0) return;
  int t = 0, tt = 0;
  for (int e = 0; e < EE; ++e) {
    offsets[e] = t;
    tileOff[e] = tt;
    cursors[e] = t;
    t += counts[e];
    tt += (counts[e] + TOK_TILE - 1) / TOK_TILE;
  }
  offsets[EE] = t;
  tileOff[EE] = tt;
  *totalTiles = tt;
}

__global__ void k_scatter(const int* __restrict__ routes, int* __restrict__ cursors,
                          int* __restrict__ tokenList) {
  int i = blockIdx.x * blockDim.x + threadIdx.x;
  if (i >= BS) return;
  int pos = atomicAdd(&cursors[routes[i]], 1);
  tokenList[pos] = i;
}

// ---------------- GEMM 1: u = Win[e] @ x ----------------

__global__ void __launch_bounds__(256) k_u_gemm(
    const bf16* __restrict__ xbf, const bf16* __restrict__ winb,
    const int* __restrict__ tokenList, const int* __restrict__ offsets,
    const int* __restrict__ tileOff, const int* __restrict__ totalTiles,
    float* __restrict__ u) {
  __shared__ bf16 xs[TOK_TILE * DD];   // 64 KB
  __shared__ int tl[TOK_TILE];
  TileInfo ti;
  if (!tile_info(offsets, tileOff, totalTiles, ti)) return;
  if (threadIdx.x < TOK_TILE) {
    int li = ti.listBase + (int)threadIdx.x;
    tl[threadIdx.x] = (li < ti.listEnd) ? tokenList[li] : -1;
  }
  __syncthreads();
  gather_rows<DD>(xbf, xs, tl);
  __syncthreads();

  const int wid = threadIdx.x >> 5;
  const int lane = threadIdx.x & 31;
  const bf16* W = winb + (size_t)ti.e * NN * DD;
  v8f acc[2][2];   // [ftile][token-half]
#pragma unroll
  for (int t = 0; t < 2; ++t)
#pragma unroll
    for (int th = 0; th < 2; ++th) acc[t][th] = vzero8();

  for (int k = 0; k < DD; k += 32) {
    v16bf b0 = load_frag(xs, DD, k, lane);
    v16bf b1 = load_frag(xs + 16 * DD, DD, k, lane);
#pragma unroll
    for (int t = 0; t < 2; ++t) {
      v16bf a = load_frag(W + (size_t)((wid * 2 + t) * 16) * DD, DD, k, lane);
      acc[t][0] = wmma_bf16(a, b0, acc[t][0]);
      acc[t][1] = wmma_bf16(a, b1, acc[t][1]);
    }
  }
  const int r = lane & 15, hf = lane >> 4;
#pragma unroll
  for (int th = 0; th < 2; ++th) {
    int tok = tl[th * 16 + r];
    if (tok < 0) continue;
    float* up = u + (size_t)tok * NN;
#pragma unroll
    for (int t = 0; t < 2; ++t) {
      int fb = (wid * 2 + t) * 16 + hf * 8;
#pragma unroll
      for (int j = 0; j < 8; ++j) up[fb + j] = acc[t][th][j];
    }
  }
}

// ---- GEMM 2 (fused): sh = silu(Wsin@x) in LDS, sel = Wsout@sh -> a, b*u, c, skip ----

__global__ void __launch_bounds__(256) k_sel_gemm(
    const bf16* __restrict__ xbf, const bf16* __restrict__ wsinb,
    const bf16* __restrict__ wsoutb, const float* __restrict__ u,
    const float* __restrict__ dparam,
    const int* __restrict__ tokenList, const int* __restrict__ offsets,
    const int* __restrict__ tileOff, const int* __restrict__ totalTiles,
    float* __restrict__ aarr, float* __restrict__ buarr,
    float* __restrict__ carr, float* __restrict__ skarr) {
  __shared__ bf16 xs[TOK_TILE * DD];   // 64 KB
  __shared__ bf16 sh[TOK_TILE * HH];   // 32 KB
  __shared__ int tl[TOK_TILE];
  TileInfo ti;
  if (!tile_info(offsets, tileOff, totalTiles, ti)) return;
  if (threadIdx.x < TOK_TILE) {
    int li = ti.listBase + (int)threadIdx.x;
    tl[threadIdx.x] = (li < ti.listEnd) ? tokenList[li] : -1;
  }
  __syncthreads();
  gather_rows<DD>(xbf, xs, tl);
  __syncthreads();

  const int wid = threadIdx.x >> 5;
  const int lane = threadIdx.x & 31;
  const int r = lane & 15, hf = lane >> 4;

  // stage 1: sh(32x512) = silu(Wsin[e](512x1024) @ x^T), 4 feature tiles per wave
  {
    const bf16* Wsi = wsinb + (size_t)ti.e * HH * DD;
    v8f acc[4][2];
#pragma unroll
    for (int t = 0; t < 4; ++t)
#pragma unroll
      for (int th = 0; th < 2; ++th) acc[t][th] = vzero8();
    for (int k = 0; k < DD; k += 32) {
      v16bf b0 = load_frag(xs, DD, k, lane);
      v16bf b1 = load_frag(xs + 16 * DD, DD, k, lane);
#pragma unroll
      for (int t = 0; t < 4; ++t) {
        v16bf a = load_frag(Wsi + (size_t)((wid * 4 + t) * 16) * DD, DD, k, lane);
        acc[t][0] = wmma_bf16(a, b0, acc[t][0]);
        acc[t][1] = wmma_bf16(a, b1, acc[t][1]);
      }
    }
#pragma unroll
    for (int th = 0; th < 2; ++th) {
      int row = th * 16 + r;   // token row in tile
#pragma unroll
      for (int t = 0; t < 4; ++t) {
        int fb = (wid * 4 + t) * 16 + hf * 8;  // 8 consecutive h-features
#pragma unroll
        for (int j = 0; j < 8; ++j) {
          float v = acc[t][th][j];
          sh[row * HH + fb + j] = (bf16)(v * sigmoidf_(v));  // silu
        }
      }
    }
  }
  __syncthreads();

  // stage 2: sel(32x1024) = Wsout[e](1024x512) @ sh^T, 8 feature tiles per wave
  {
    const bf16* Wso = wsoutb + (size_t)ti.e * (4 * NN) * HH;
    v8f acc[8][2];
#pragma unroll
    for (int t = 0; t < 8; ++t)
#pragma unroll
      for (int th = 0; th < 2; ++th) acc[t][th] = vzero8();
    for (int k = 0; k < HH; k += 32) {
      v16bf b0 = load_frag(sh, HH, k, lane);
      v16bf b1 = load_frag(sh + 16 * HH, HH, k, lane);
#pragma unroll
      for (int t = 0; t < 8; ++t) {
        v16bf a = load_frag(Wso + (size_t)((wid * 8 + t) * 16) * HH, HH, k, lane);
        acc[t][0] = wmma_bf16(a, b0, acc[t][0]);
        acc[t][1] = wmma_bf16(a, b1, acc[t][1]);
      }
    }
#pragma unroll
    for (int th = 0; th < 2; ++th) {
      int tok = tl[th * 16 + r];
      if (tok < 0) continue;
      const float* up = u + (size_t)tok * NN;
#pragma unroll
      for (int t = 0; t < 8; ++t) {
        int ob = (wid * 8 + t) * 16 + hf * 8;  // global sel-feature base for this thread
        int seg = ob >> 8;                     // 0:a 1:b 2:c 3:d
        int nb = ob & 255;
        size_t base = (size_t)tok * NN + nb;
#pragma unroll
        for (int j = 0; j < 8; ++j) {
          float v = acc[t][th][j];
          if (seg == 0)      aarr[base + j] = sigmoidf_(v);
          else if (seg == 1) buarr[base + j] = tanhf(v) * up[nb + j];
          else if (seg == 2) carr[base + j] = tanhf(v);
          else               skarr[base + j] = dparam[ti.e * NN + nb + j] * sigmoidf_(v) * up[nb + j];
        }
      }
    }
  }
}

// ---------------- sequential scan: h = a*h + b*u ; y = c*h + skip ----------------

__global__ void __launch_bounds__(NN) k_scan(
    const int* __restrict__ routes, const float* __restrict__ a,
    const float* __restrict__ bu, const float* __restrict__ c,
    const float* __restrict__ sk, bf16* __restrict__ ybf) {
  int b = blockIdx.x;       // one block per batch row (blockDim == NN)
  int n = threadIdx.x;
  float h[EE];
#pragma unroll
  for (int e = 0; e < EE; ++e) h[e] = 0.f;
  for (int t = 0; t < SS; ++t) {
    int base = b * SS + t;
    int r = routes[base];
    size_t i = (size_t)base * NN + n;
    float av = a[i], buv = bu[i], cv = c[i], skv = sk[i];
    float hr = 0.f;
#pragma unroll
    for (int e = 0; e < EE; ++e) hr = (r == e) ? h[e] : hr;
    float hn = av * hr + buv;
#pragma unroll
    for (int e = 0; e < EE; ++e) h[e] = (r == e) ? hn : h[e];
    ybf[i] = (bf16)(cv * hn + skv);
  }
}

// ---------------- GEMM 3: out = Wout[e] @ y ----------------

__global__ void __launch_bounds__(256) k_out_gemm(
    const bf16* __restrict__ ybf, const bf16* __restrict__ woutb,
    const int* __restrict__ tokenList, const int* __restrict__ offsets,
    const int* __restrict__ tileOff, const int* __restrict__ totalTiles,
    float* __restrict__ out) {
  __shared__ bf16 ys[TOK_TILE * NN];   // 16 KB
  __shared__ int tl[TOK_TILE];
  TileInfo ti;
  if (!tile_info(offsets, tileOff, totalTiles, ti)) return;
  if (threadIdx.x < TOK_TILE) {
    int li = ti.listBase + (int)threadIdx.x;
    tl[threadIdx.x] = (li < ti.listEnd) ? tokenList[li] : -1;
  }
  __syncthreads();
  gather_rows<NN>(ybf, ys, tl);
  __syncthreads();

  const int wid = threadIdx.x >> 5;
  const int lane = threadIdx.x & 31;
  const bf16* W = woutb + (size_t)ti.e * DD * NN;
  v8f acc[8][2];
#pragma unroll
  for (int t = 0; t < 8; ++t)
#pragma unroll
    for (int th = 0; th < 2; ++th) acc[t][th] = vzero8();
  for (int k = 0; k < NN; k += 32) {
    v16bf b0 = load_frag(ys, NN, k, lane);
    v16bf b1 = load_frag(ys + 16 * NN, NN, k, lane);
#pragma unroll
    for (int t = 0; t < 8; ++t) {
      v16bf a = load_frag(W + (size_t)((wid * 8 + t) * 16) * NN, NN, k, lane);
      acc[t][0] = wmma_bf16(a, b0, acc[t][0]);
      acc[t][1] = wmma_bf16(a, b1, acc[t][1]);
    }
  }
  const int r = lane & 15, hf = lane >> 4;
#pragma unroll
  for (int th = 0; th < 2; ++th) {
    int tok = tl[th * 16 + r];
    if (tok < 0) continue;
    float* op = out + (size_t)tok * DD;
#pragma unroll
    for (int t = 0; t < 8; ++t) {
      int fb = (wid * 8 + t) * 16 + hf * 8;
#pragma unroll
      for (int j = 0; j < 8; ++j) op[fb + j] = acc[t][th][j];
    }
  }
}

// ---------------- host launcher ----------------

extern "C" void kernel_launch(void* const* d_in, const int* in_sizes, int n_in,
                              void* d_out, int out_size, void* d_ws, size_t ws_size,
                              hipStream_t stream) {
  (void)in_sizes; (void)n_in; (void)out_size; (void)ws_size;
  const float* x      = (const float*)d_in[0];
  const int*   tokid  = (const int*)d_in[1];
  const float* Win    = (const float*)d_in[2];
  const float* Wsin   = (const float*)d_in[3];
  const float* Wsout  = (const float*)d_in[4];
  const float* Wout   = (const float*)d_in[5];
  const float* dparam = (const float*)d_in[6];
  float* out = (float*)d_out;

  char* p = (char*)d_ws;
  auto take = [&](size_t bytes) -> char* {
    char* q = p;
    p += (bytes + 255) & ~(size_t)255;
    return q;
  };

  int* routes     = (int*)take(sizeof(int) * BS);
  int* counts     = (int*)take(sizeof(int) * EE);
  int* offsets    = (int*)take(sizeof(int) * (EE + 1));
  int* tileOff    = (int*)take(sizeof(int) * (EE + 1));
  int* cursors    = (int*)take(sizeof(int) * EE);
  int* totalTiles = (int*)take(sizeof(int));
  int* tokenList  = (int*)take(sizeof(int) * BS);
  bf16* xbf    = (bf16*)take(sizeof(bf16) * (size_t)BS * DD);
  bf16* winb   = (bf16*)take(sizeof(bf16) * (size_t)EE * NN * DD);
  bf16* wsinb  = (bf16*)take(sizeof(bf16) * (size_t)EE * HH * DD);
  bf16* wsoutb = (bf16*)take(sizeof(bf16) * (size_t)EE * 4 * NN * HH);
  bf16* woutb  = (bf16*)take(sizeof(bf16) * (size_t)EE * DD * NN);
  float* u     = (float*)take(sizeof(float) * (size_t)BS * NN);
  float* aarr  = (float*)take(sizeof(float) * (size_t)BS * NN);
  float* buarr = (float*)take(sizeof(float) * (size_t)BS * NN);
  float* carr  = (float*)take(sizeof(float) * (size_t)BS * NN);
  float* skarr = (float*)take(sizeof(float) * (size_t)BS * NN);
  bf16* ybf    = (bf16*)take(sizeof(bf16) * (size_t)BS * NN);

  k_init<<<1, 64, 0, stream>>>(counts);

  int n;
  n = BS * DD;          k_f2bf<<<(n + 255) / 256, 256, 0, stream>>>(x, xbf, n);
  n = EE * NN * DD;     k_f2bf<<<(n + 255) / 256, 256, 0, stream>>>(Win, winb, n);
  n = EE * HH * DD;     k_f2bf<<<(n + 255) / 256, 256, 0, stream>>>(Wsin, wsinb, n);
  n = EE * 4 * NN * HH; k_f2bf<<<(n + 255) / 256, 256, 0, stream>>>(Wsout, wsoutb, n);
  n = EE * DD * NN;     k_f2bf<<<(n + 255) / 256, 256, 0, stream>>>(Wout, woutb, n);

  k_route<<<BS / 256, 256, 0, stream>>>(tokid, routes, counts);
  k_offsets<<<1, 1, 0, stream>>>(counts, offsets, tileOff, cursors, totalTiles);
  k_scatter<<<BS / 256, 256, 0, stream>>>(routes, cursors, tokenList);

  k_u_gemm<<<TILE_MAX, 256, 0, stream>>>(xbf, winb, tokenList, offsets, tileOff,
                                         totalTiles, u);
  k_sel_gemm<<<TILE_MAX, 256, 0, stream>>>(xbf, wsinb, wsoutb, u, dparam, tokenList,
                                           offsets, tileOff, totalTiles,
                                           aarr, buarr, carr, skarr);
  k_scan<<<BB, NN, 0, stream>>>(routes, aarr, buarr, carr, skarr, ybf);
  k_out_gemm<<<TILE_MAX, 256, 0, stream>>>(ybf, woutb, tokenList, offsets, tileOff,
                                           totalTiles, out);
}
